// MemoryUnit_69483980915407
// MI455X (gfx1250) — compile-verified, matrix operand order checked by reading, not run on hardware
//
#include <hip/hip_runtime.h>
#include <hip/hip_bf16.h>
#include <math.h>

#define MEM_DIM 4000
#define FEAT    2048
#define BATCH   8192
#define NTILES  (MEM_DIM / 16)   // 250
#define THRESH  (1.0f / 4000.0f)
#define EPSV    1e-12f
#define COS_EPS 1e-8f

typedef __attribute__((ext_vector_type(16))) __bf16 v16bf;
typedef __attribute__((ext_vector_type(8)))  float  v8f;

union FragBF { v16bf v; uint4 q[2]; };

__device__ __forceinline__ unsigned short f32_to_bf16_rn(float f) {
  unsigned int u = __float_as_uint(f);
  u += 0x7FFFu + ((u >> 16) & 1u);     // round-to-nearest-even
  return (unsigned short)(u >> 16);
}

__device__ __forceinline__ void load_frag(FragBF& f, const unsigned short* p) {
  f.q[0] = *(const uint4*)(p);
  f.q[1] = *(const uint4*)(p + 16);
}

#define WMMA_BF16(A, B, C) \
  __builtin_amdgcn_wmma_f32_16x16x32_bf16(false, (A).v, false, (B).v, (short)0, (C), false, false)

// ---------------------------------------------------------------------------
// Prep: x -> bf16 row-major, row norms
// ---------------------------------------------------------------------------
__global__ __launch_bounds__(256)
void prep_x_kernel(const float* __restrict__ x,
                   unsigned short* __restrict__ x16,
                   float* __restrict__ xnorm) {
  const int r = blockIdx.x;
  const int t = threadIdx.x;
  const float* xr = x + (size_t)r * FEAT;
  unsigned short* dr = x16 + (size_t)r * FEAT;
  float s = 0.f;
  for (int f = t; f < FEAT; f += 256) {
    float v = xr[f];
    s += v * v;
    dr[f] = f32_to_bf16_rn(v);
  }
  __shared__ float red[256];
  red[t] = s; __syncthreads();
  for (int w = 128; w > 0; w >>= 1) {
    if (t < w) red[t] += red[t + w];
    __syncthreads();
  }
  if (t == 0) xnorm[r] = sqrtf(red[0]);
}

// ---------------------------------------------------------------------------
// Prep: memory -> bf16 row-major (GEMM1 B) + bf16 transposed (GEMM2 B), norms
// ---------------------------------------------------------------------------
__global__ __launch_bounds__(256)
void prep_mem_kernel(const float* __restrict__ mem,
                     unsigned short* __restrict__ m16,
                     unsigned short* __restrict__ mT16,
                     float* __restrict__ mnorm) {
  const int r = blockIdx.x;
  const int t = threadIdx.x;
  const float* mr = mem + (size_t)r * FEAT;
  unsigned short* dr = m16 + (size_t)r * FEAT;
  float s = 0.f;
  for (int f = t; f < FEAT; f += 256) {
    float v = mr[f];
    s += v * v;
    unsigned short b = f32_to_bf16_rn(v);
    dr[f] = b;
    mT16[(size_t)f * MEM_DIM + r] = b;   // transposed copy for GEMM2 (merged in L2)
  }
  __shared__ float red[256];
  red[t] = s; __syncthreads();
  for (int w = 128; w > 0; w >>= 1) {
    if (t < w) red[t] += red[t + w];
    __syncthreads();
  }
  if (t == 0) mnorm[r] = sqrtf(red[0]);
}

// ---------------------------------------------------------------------------
// Fused: GEMM1 (bf16 WMMA) -> softmax/shrink/L1 in LDS -> GEMM2 (bf16 WMMA)
// One workgroup (8 wave32s) owns 16 rows of x. Dynamic LDS:
//   float logits[16*4000] | float red[256] | float rowv[16] | float xn_s[16]
// ---------------------------------------------------------------------------
__global__ __launch_bounds__(256)
void fused_kernel(const unsigned short* __restrict__ x16,
                  const unsigned short* __restrict__ m16,
                  const unsigned short* __restrict__ mT16,
                  const float* __restrict__ xnorm,
                  const float* __restrict__ mnorm,
                  float* __restrict__ out) {
  extern __shared__ float smem[];
  float* logits = smem;                       // 64000 floats (256 KB)
  float* red    = smem + 16 * MEM_DIM;        // 256
  float* rowv   = red + 256;                  // 16
  float* xn_s   = rowv + 16;                  // 16

  const int t        = threadIdx.x;
  const int lane     = t & 31;
  const int wave     = t >> 5;
  const int row_base = blockIdx.x * 16;

  if (t < 16) xn_s[t] = xnorm[row_base + t];
  __syncthreads();

  // ---------------- Phase 1: logits = (x @ mem^T) / denom -----------------
  {
    const int rc    = lane & 15;             // A row (M) / B column (N) index
    const int khalf = (lane >> 4) * 8;       // lanes 16..31 hold K+8 half
    const int moff  = (lane >> 4) * 8;
    const unsigned short* aBase = x16 + (size_t)(row_base + rc) * FEAT + khalf;

    // contiguous tile split: waves 0,1 -> 32 tiles, waves 2..7 -> 31 tiles
    const int start = wave * 31 + (wave < 2 ? wave : 2);
    const int end   = start + 31 + (wave < 2 ? 1 : 0);

    int nt = start;
    // ---- groups of 4 N-tiles, K ping-pong double buffered ----
    #pragma unroll 1
    for (; nt + 4 <= end; nt += 4) {
      const unsigned short* bB0 = m16 + (size_t)((nt + 0) * 16 + rc) * FEAT + khalf;
      const unsigned short* bB1 = m16 + (size_t)((nt + 1) * 16 + rc) * FEAT + khalf;
      const unsigned short* bB2 = m16 + (size_t)((nt + 2) * 16 + rc) * FEAT + khalf;
      const unsigned short* bB3 = m16 + (size_t)((nt + 3) * 16 + rc) * FEAT + khalf;

      v8f acc0 = {}, acc1 = {}, acc2 = {}, acc3 = {};
      FragBF a0, b00, b01, b02, b03;         // set 0
      FragBF a1, b10, b11, b12, b13;         // set 1

      load_frag(a0,  aBase);
      load_frag(b00, bB0); load_frag(b01, bB1);
      load_frag(b02, bB2); load_frag(b03, bB3);

      #pragma unroll 1
      for (int kb = 0; kb < FEAT - 64; kb += 64) {
        load_frag(a1,  aBase + kb + 32);
        load_frag(b10, bB0 + kb + 32); load_frag(b11, bB1 + kb + 32);
        load_frag(b12, bB2 + kb + 32); load_frag(b13, bB3 + kb + 32);
        acc0 = WMMA_BF16(a0, b00, acc0);
        acc1 = WMMA_BF16(a0, b01, acc1);
        acc2 = WMMA_BF16(a0, b02, acc2);
        acc3 = WMMA_BF16(a0, b03, acc3);
        load_frag(a0,  aBase + kb + 64);
        load_frag(b00, bB0 + kb + 64); load_frag(b01, bB1 + kb + 64);
        load_frag(b02, bB2 + kb + 64); load_frag(b03, bB3 + kb + 64);
        acc0 = WMMA_BF16(a1, b10, acc0);
        acc1 = WMMA_BF16(a1, b11, acc1);
        acc2 = WMMA_BF16(a1, b12, acc2);
        acc3 = WMMA_BF16(a1, b13, acc3);
      }
      // tail: set0 holds kb=FEAT-64, load+do kb=FEAT-32
      load_frag(a1,  aBase + FEAT - 32);
      load_frag(b10, bB0 + FEAT - 32); load_frag(b11, bB1 + FEAT - 32);
      load_frag(b12, bB2 + FEAT - 32); load_frag(b13, bB3 + FEAT - 32);
      acc0 = WMMA_BF16(a0, b00, acc0);
      acc1 = WMMA_BF16(a0, b01, acc1);
      acc2 = WMMA_BF16(a0, b02, acc2);
      acc3 = WMMA_BF16(a0, b03, acc3);
      acc0 = WMMA_BF16(a1, b10, acc0);
      acc1 = WMMA_BF16(a1, b11, acc1);
      acc2 = WMMA_BF16(a1, b12, acc2);
      acc3 = WMMA_BF16(a1, b13, acc3);

      #pragma unroll
      for (int g = 0; g < 4; ++g) {
        const v8f acc = (g == 0) ? acc0 : (g == 1) ? acc1 : (g == 2) ? acc2 : acc3;
        const int ncol = (nt + g) * 16 + rc;
        const float mn = mnorm[ncol];
        #pragma unroll
        for (int v = 0; v < 8; ++v) {
          const int m = v + moff;
          const float sc = __builtin_amdgcn_rcpf(fmaxf(xn_s[m] * mn, COS_EPS));
          logits[m * MEM_DIM + ncol] = acc[v] * sc;
        }
      }
    }
    // ---- remainder tiles (0..3), simple loop ----
    #pragma unroll 1
    for (; nt < end; ++nt) {
      const unsigned short* bBase = m16 + (size_t)(nt * 16 + rc) * FEAT + khalf;
      v8f acc = {};
      #pragma unroll 1
      for (int kb = 0; kb < FEAT; kb += 32) {
        FragBF a, b;
        load_frag(a, aBase + kb);
        load_frag(b, bBase + kb);
        acc = WMMA_BF16(a, b, acc);
      }
      const int ncol = nt * 16 + rc;
      const float mn = mnorm[ncol];
      #pragma unroll
      for (int v = 0; v < 8; ++v) {
        const int m = v + moff;
        const float sc = __builtin_amdgcn_rcpf(fmaxf(xn_s[m] * mn, COS_EPS));
        logits[m * MEM_DIM + ncol] = acc[v] * sc;
      }
    }
  }
  __syncthreads();

  // ---------------- Phase 2: softmax + shrinkage + L1 norm ----------------
  const int m = t >> 4;                       // 16 threads per row
  const int j = t & 15;

  // row max
  float pmax = -3.4e38f;
  for (int n = j; n < MEM_DIM; n += 16) pmax = fmaxf(pmax, logits[m * MEM_DIM + n]);
  red[t] = pmax; __syncthreads();
  if (j == 0) {
    float v = red[t];
    for (int i = 1; i < 16; ++i) v = fmaxf(v, red[t + i]);
    rowv[m] = v;
  }
  __syncthreads();
  const float rmax = rowv[m];

  // exp + row sum
  float psum = 0.f;
  for (int n = j; n < MEM_DIM; n += 16) {
    float e = __expf(logits[m * MEM_DIM + n] - rmax);
    logits[m * MEM_DIM + n] = e;
    psum += e;
  }
  red[t] = psum; __syncthreads();
  if (j == 0) {
    float v = 0.f;
    for (int i = 0; i < 16; ++i) v += red[t + i];
    rowv[m] = v;
  }
  __syncthreads();
  const float rinv = __builtin_amdgcn_rcpf(rowv[m]);

  // shrinkage + abs sum (rcp approx instead of IEEE divide)
  float pabs = 0.f;
  for (int n = j; n < MEM_DIM; n += 16) {
    float w  = logits[m * MEM_DIM + n] * rinv;
    float d  = w - THRESH;
    float w2 = fmaxf(d, 0.f) * w * __builtin_amdgcn_rcpf(fabsf(d) + EPSV);
    logits[m * MEM_DIM + n] = w2;
    pabs += fabsf(w2);
  }
  red[t] = pabs; __syncthreads();
  if (j == 0) {
    float v = 0.f;
    for (int i = 0; i < 16; ++i) v += red[t + i];
    rowv[m] = __builtin_amdgcn_rcpf(fmaxf(v, EPSV));  // fold L1 scale into pack
  }
  __syncthreads();

  // ---- in-place repack: f32 w -> bf16 into low half of LDS (chunked) ----
  unsigned short* wlds = (unsigned short*)smem;
  #pragma unroll 1
  for (int c = 0; c < 32; ++c) {
    const int base = c * 2048 + t * 8;
    float tmp[8];
    #pragma unroll
    for (int e = 0; e < 8; ++e) {
      const int idx = base + e;
      tmp[e] = (idx < 16 * MEM_DIM) ? logits[idx] * rowv[idx / MEM_DIM] : 0.f;
    }
    __syncthreads();
    #pragma unroll
    for (int e = 0; e < 8; ++e) {
      const int idx = base + e;
      if (idx < 16 * MEM_DIM) wlds[idx] = f32_to_bf16_rn(tmp[e]);
    }
    __syncthreads();
  }

  // ---------------- Phase 3: out = w @ mem (A from LDS, B from mT16) ------
  {
    const int rc    = lane & 15;
    const int khalf = (lane >> 4) * 8;
    const int n0    = wave * 256;             // 256 output columns per wave
    v8f acc[16];
    #pragma unroll
    for (int i = 0; i < 16; ++i) acc[i] = (v8f){};

    const unsigned short* aLds  = wlds + (size_t)rc * MEM_DIM + khalf;
    const unsigned short* bBase = mT16 + (size_t)(n0 + rc) * MEM_DIM + khalf;

    #pragma unroll 1
    for (int kb = 0; kb < MEM_DIM; kb += 32) {
      FragBF a;
      load_frag(a, aLds + kb);                // ds_load_b128 x2
      #pragma unroll
      for (int ntile = 0; ntile < 16; ++ntile) {
        FragBF b;
        load_frag(b, bBase + (size_t)ntile * 16 * MEM_DIM + kb);
        acc[ntile] = WMMA_BF16(a, b, acc[ntile]);
      }
    }

    const int moff = (lane >> 4) * 8;
    #pragma unroll
    for (int ntile = 0; ntile < 16; ++ntile) {
      const int n = n0 + ntile * 16 + rc;
      #pragma unroll
      for (int v = 0; v < 8; ++v) {
        const int mm = v + moff;
        out[(size_t)(row_base + mm) * FEAT + n] = acc[ntile][v];
      }
    }
  }
}

// ---------------------------------------------------------------------------
extern "C" void kernel_launch(void* const* d_in, const int* in_sizes, int n_in,
                              void* d_out, int out_size, void* d_ws, size_t ws_size,
                              hipStream_t stream) {
  (void)in_sizes; (void)n_in; (void)out_size; (void)ws_size;
  const float* x   = (const float*)d_in[0];   // [8192, 2048]
  const float* mem = (const float*)d_in[1];   // [4000, 2048]
  float* out = (float*)d_out;                 // [8192, 2048]

  char* ws = (char*)d_ws;
  size_t off = 0;
  unsigned short* x16  = (unsigned short*)(ws + off); off += (size_t)BATCH  * FEAT * 2;
  unsigned short* m16  = (unsigned short*)(ws + off); off += (size_t)MEM_DIM* FEAT * 2;
  unsigned short* mT16 = (unsigned short*)(ws + off); off += (size_t)FEAT   * MEM_DIM * 2;
  float* xnorm = (float*)(ws + off); off += (size_t)BATCH   * 4;
  float* mnorm = (float*)(ws + off); off += (size_t)MEM_DIM * 4;

  prep_x_kernel  <<<BATCH,   256, 0, stream>>>(x,   x16, xnorm);
  prep_mem_kernel<<<MEM_DIM, 256, 0, stream>>>(mem, m16, mT16, mnorm);

  const size_t shbytes = (size_t)(16 * MEM_DIM + 256 + 16 + 16) * sizeof(float); // ~251 KB
  fused_kernel<<<BATCH / 16, 256, shbytes, stream>>>(x16, m16, mT16, xnorm, mnorm, out);
}
